// Net_1700807049793
// MI455X (gfx1250) — matrix operand order repaired and checked by reading, not compiled, not run on hardware
//
#include <hip/hip_runtime.h>

typedef _Float16 half_t;
typedef __attribute__((ext_vector_type(16))) _Float16 v16h;
typedef __attribute__((ext_vector_type(8)))  _Float16 v8h;
typedef __attribute__((ext_vector_type(8)))  float    v8f;
typedef int v4i_ __attribute__((vector_size(16)));

#define NEGV (-1000000000.0f)

// Problem constants (from reference)
#define VNUM 30000
#define EDIM 300
#define EPAD 320
#define HDIM 512
#define H3   1536
#define BN   64
#define LQ   64
#define AN   7
#define LO   16
#define SN   50
#define LS   32
#define WN   3
#define SPN  150            // SN*WN
#define BSN  9600           // BN*SPN
#define NTOK 113664         // 4096 + 7168 + 102400
#define ROWO 4096
#define ROWC 11264
#define NSEQ 3712           // 64 + 448 + 3200

#if __has_builtin(__builtin_amdgcn_global_load_async_to_lds_b128)
#define HAVE_ASYNC_LDS 1
#endif

__device__ __forceinline__ void async_wait0() {
#ifdef HAVE_ASYNC_LDS
#if __has_builtin(__builtin_amdgcn_s_wait_asynccnt)
    __builtin_amdgcn_s_wait_asynccnt(0);
#else
    asm volatile("s_wait_asynccnt 0x0" ::: "memory");
#endif
#endif
}

__device__ __forceinline__ void lds_stage_b128(const half_t* src, half_t* dst) {
#ifdef HAVE_ASYNC_LDS
    __builtin_amdgcn_global_load_async_to_lds_b128(
        (__attribute__((address_space(1))) v4i_*)(void*)src,
        (__attribute__((address_space(3))) v4i_*)(void*)dst, 0, 0);
#else
    *(v8h*)dst = *(const v8h*)src;
#endif
}

__device__ __forceinline__ float sigm(float x) { return 1.f / (1.f + expf(-x)); }
__device__ __forceinline__ float entf(float p) { return -p * log2f(p) - (1.f - p) * log2f(1.f - p); }

// ---------------------------------------------------------------------------
// Pack an f32 weight matrix W (K x N logical, optionally given transposed as
// N x K) into f16 WMMA B-fragment order:
//   index = (((nt*KK + kk)*32 + lane)*16 + e)
//   lane 0-15  -> k = kk*32 + (e&7) + (e>>3)*16       col n = nt*16 + (lane&15)
//   lane 16-31 -> k = kk*32 + 8 + (e&7) + (e>>3)*16
// ---------------------------------------------------------------------------
__global__ void pack_b_kernel(const float* __restrict__ W, half_t* __restrict__ Bf,
                              int K, int Kp, int N, int transposed) {
    size_t id = (size_t)blockIdx.x * blockDim.x + threadIdx.x;
    int KK = Kp >> 5;
    size_t total = (size_t)(N >> 4) * KK * 512;
    if (id >= total) return;
    int e    = (int)(id & 15);
    int lane = (int)((id >> 4) & 31);
    size_t t2 = id >> 9;
    int kk = (int)(t2 % KK);
    int nt = (int)(t2 / KK);
    int k = (kk << 5) + ((e >> 3) << 4) + ((lane >> 4) << 3) + (e & 7);
    int n = (nt << 4) + (lane & 15);
    float v = 0.f;
    if (k < K) v = transposed ? W[(size_t)n * K + k] : W[(size_t)k * N + n];
    Bf[id] = (half_t)v;
}

// ---------------------------------------------------------------------------
// Gather embeddings for all tokens into X [NTOK, EPAD] f16 (zero padded).
// ---------------------------------------------------------------------------
__global__ void gather_embed_kernel(const int* __restrict__ que_ix,
                                    const int* __restrict__ opt_ix,
                                    const int* __restrict__ cp_ix,
                                    const float* __restrict__ emb,
                                    half_t* __restrict__ X) {
    size_t id = (size_t)blockIdx.x * blockDim.x + threadIdx.x;
    if (id >= (size_t)NTOK * EPAD) return;
    int c = (int)(id % EPAD);
    size_t row = id / EPAD;
    int tok;
    if (row < ROWO)      tok = que_ix[row];
    else if (row < ROWC) tok = opt_ix[row - ROWO];
    else                 tok = cp_ix[row - ROWC];
    float v = (c < EDIM) ? emb[(size_t)tok * EDIM + c] : 0.f;
    X[id] = (half_t)v;
}

// ---------------------------------------------------------------------------
// WMMA GEMM. One block (8 waves) computes a 128x16 tile: 8 m-tiles sharing a
// single n-tile. The shared B fragments stream through LDS in 4-kstep chunks,
// double buffered, staged with async global->LDS loads (each wave issues its
// own pieces and waits its own ASYNCcnt before the block barrier).
// ST: 0 = store f16, 1 = store f32, 2 = store both, 3 = score epilogue
//     (t = acc+bias, optional relu, dot with w2[n], atomicAdd into outS[row])
// ---------------------------------------------------------------------------
template <int ST, bool RELU>
__global__ void gemm_wmma(const half_t* __restrict__ A, int lda,
                          const half_t* __restrict__ Bf,
                          const float* __restrict__ bias,
                          int M, int N, int Kp,
                          float* __restrict__ outF, half_t* __restrict__ outH,
                          const float* __restrict__ w2, float* __restrict__ outS) {
    __shared__ __align__(32) half_t bsh[2][2048];   // 2 x (4 ksteps x 512 halfs)

    int tid  = threadIdx.x;
    int lane = tid & 31;
    int wave = tid >> 5;
    int tilesN = N >> 4;
    int tilesM = M >> 4;
    int nt  = blockIdx.x % tilesN;
    int mtg = blockIdx.x / tilesN;
    int mt  = mtg * 8 + wave;
    bool valid = (mt < tilesM);
    int mtc = valid ? mt : (tilesM - 1);
    int KK = Kp >> 5;
    int NC = (KK + 3) >> 2;

    int mrow = (mtc << 4) + (lane & 15);
    int kb = (lane & 16) ? 8 : 0;
    const half_t* arow  = A + (size_t)mrow * lda + kb;
    const half_t* bbase = Bf + (((size_t)nt * KK) << 9);

    // stage chunk c into buffer buf (256 threads x 16B pieces)
    auto stage = [&](int c, int buf) {
        int cnt = KK - c * 4; if (cnt > 4) cnt = 4;
        int pieces = cnt << 6;              // cnt * 512 halfs / 8
        if (tid < pieces)
            lds_stage_b128(bbase + (((size_t)c * 4) << 9) + ((size_t)tid << 3),
                           &bsh[buf][tid << 3]);
    };

    stage(0, 0);
    async_wait0();
    __syncthreads();

    v8f ac0 = {}, ac1 = {}, ac2 = {}, ac3 = {};
    for (int c = 0; c < NC; ++c) {
        int buf = c & 1;
        if (c + 1 < NC) stage(c + 1, buf ^ 1);
        int cnt = KK - c * 4; if (cnt > 4) cnt = 4;
#pragma unroll
        for (int kl = 0; kl < 4; ++kl) {
            if (kl >= cnt) break;
            int kk = c * 4 + kl;
            v8h lo = *(const v8h*)(arow + (kk << 5));
            v8h hi = *(const v8h*)(arow + (kk << 5) + 16);
            v16h af;
#pragma unroll
            for (int e2 = 0; e2 < 8; ++e2) { af[e2] = lo[e2]; af[e2 + 8] = hi[e2]; }
            v16h bfrag = *(const v16h*)(&bsh[buf][(kl << 9) + (lane << 4)]);
            if (kl == 0)      ac0 = __builtin_amdgcn_wmma_f32_16x16x32_f16(false, af, false, bfrag, (short)0, ac0, false, false);
            else if (kl == 1) ac1 = __builtin_amdgcn_wmma_f32_16x16x32_f16(false, af, false, bfrag, (short)0, ac1, false, false);
            else if (kl == 2) ac2 = __builtin_amdgcn_wmma_f32_16x16x32_f16(false, af, false, bfrag, (short)0, ac2, false, false);
            else              ac3 = __builtin_amdgcn_wmma_f32_16x16x32_f16(false, af, false, bfrag, (short)0, ac3, false, false);
        }
        async_wait0();
        __syncthreads();
    }
    v8f acc = (ac0 + ac1) + (ac2 + ac3);

    int n = (nt << 4) + (lane & 15);
    int rbase = (mtc << 4) + ((lane >> 4) << 3);
    float bv = bias ? bias[n] : 0.f;

    if (ST == 3) {
        float wn = w2[n];
        float sred[8];
#pragma unroll
        for (int r = 0; r < 8; ++r) {
            float tv = acc[r] + bv;
            if (RELU) tv = fmaxf(tv, 0.f);
            sred[r] = tv * wn;
        }
        for (int mk = 1; mk < 16; mk <<= 1)
#pragma unroll
            for (int r = 0; r < 8; ++r) sred[r] += __shfl_xor(sred[r], mk, 32);
        if (valid && (lane & 15) == 0)
#pragma unroll
            for (int r = 0; r < 8; ++r) atomicAdd(outS + rbase + r, sred[r]);
    } else if (valid) {
#pragma unroll
        for (int r = 0; r < 8; ++r) {
            float tv = acc[r] + bv;
            if (RELU) tv = fmaxf(tv, 0.f);
            size_t idx = (size_t)(rbase + r) * N + n;
            if (ST == 1 || ST == 2) outF[idx] = tv;
            if (ST == 0 || ST == 2) outH[idx] = (half_t)tv;
        }
    }
}

// ---------------------------------------------------------------------------
// GRU recurrent scan. One workgroup (256 threads / 8 waves) owns 32 sequences.
// h kept in LDS in A-fragment layout, double buffered (2 x 32KB).
// Per step each wave does 8 (m-tile, j-tile) tasks: 3 WMMA gate column tiles.
// Accumulators are split even/odd-k (6 sets) so dependent WMMAs sit >=5 apart
// (f16 WMMA->WMMA RAW needs 1 NOP + 4 coexec slots per ISA 7.12.1).
// ---------------------------------------------------------------------------
__global__ void gru_scan_kernel(const half_t* __restrict__ XG,
                                const half_t* __restrict__ WhF,
                                const float* __restrict__ bh,
                                half_t* __restrict__ Hseq) {
    extern __shared__ __attribute__((aligned(32))) half_t hbuf[]; // 32768 halfs
    half_t* buf0 = hbuf;
    half_t* buf1 = hbuf + 16384;

    int bid = blockIdx.x;
    int T, wgRowBase;
    if (bid < 2)       { T = 64; wgRowBase = bid * 2048; }
    else if (bid < 16) { T = 16; wgRowBase = ROWO + (bid - 2) * 512; }
    else               { T = 32; wgRowBase = ROWC + (bid - 16) * 1024; }

    int tid  = threadIdx.x;
    int lane = tid & 31;
    int wave = tid >> 5;

    for (int i = tid; i < 32768; i += 256) hbuf[i] = (half_t)0.f;
    __syncthreads();

    for (int t = 0; t < T; ++t) {
        half_t* cur = (t & 1) ? buf1 : buf0;
        half_t* nxt = (t & 1) ? buf0 : buf1;
        for (int i = 0; i < 8; ++i) {
            int tt = wave * 8 + i;
            int mt = tt >> 5;       // 0..1
            int jt = tt & 31;       // 0..31 : h-dim tile
            v8f aR0 = {}, aZ0 = {}, aN0 = {}, aR1 = {}, aZ1 = {}, aN1 = {};
            for (int kk2 = 0; kk2 < 8; ++kk2) {
                int kkA = kk2 * 2, kkB = kkA + 1;
                v16h afA = *(const v16h*)(cur + ((((mt * 16 + kkA) * 32) + lane) << 4));
                v16h bRA = *(const v16h*)(WhF + ((((size_t)(jt)      * 16 + kkA) * 32 + lane) << 4));
                v16h bZA = *(const v16h*)(WhF + ((((size_t)(32 + jt) * 16 + kkA) * 32 + lane) << 4));
                v16h bNA = *(const v16h*)(WhF + ((((size_t)(64 + jt) * 16 + kkA) * 32 + lane) << 4));
                aR0 = __builtin_amdgcn_wmma_f32_16x16x32_f16(false, afA, false, bRA, (short)0, aR0, false, false);
                aZ0 = __builtin_amdgcn_wmma_f32_16x16x32_f16(false, afA, false, bZA, (short)0, aZ0, false, false);
                aN0 = __builtin_amdgcn_wmma_f32_16x16x32_f16(false, afA, false, bNA, (short)0, aN0, false, false);
                v16h afB = *(const v16h*)(cur + ((((mt * 16 + kkB) * 32) + lane) << 4));
                v16h bRB = *(const v16h*)(WhF + ((((size_t)(jt)      * 16 + kkB) * 32 + lane) << 4));
                v16h bZB = *(const v16h*)(WhF + ((((size_t)(32 + jt) * 16 + kkB) * 32 + lane) << 4));
                v16h bNB = *(const v16h*)(WhF + ((((size_t)(64 + jt) * 16 + kkB) * 32 + lane) << 4));
                aR1 = __builtin_amdgcn_wmma_f32_16x16x32_f16(false, afB, false, bRB, (short)0, aR1, false, false);
                aZ1 = __builtin_amdgcn_wmma_f32_16x16x32_f16(false, afB, false, bZB, (short)0, aZ1, false, false);
                aN1 = __builtin_amdgcn_wmma_f32_16x16x32_f16(false, afB, false, bNB, (short)0, aN1, false, false);
            }
            int j   = (jt << 4) + (lane & 15);
            int kk2i = j >> 5;
            int kl  = j & 31;
            int lp  = ((kl >> 3) & 1) << 4;
            int e   = (kl & 7) + (((kl >> 4) & 1) << 3);
            float bhr = bh[j], bhz = bh[512 + j], bhn = bh[1024 + j];
#pragma unroll
            for (int r = 0; r < 8; ++r) {
                int m = (mt << 4) + ((lane >> 4) << 3) + r;
                size_t row = (size_t)wgRowBase + (size_t)m * T + t;
                float xr = (float)XG[row * H3 + j];
                float xz = (float)XG[row * H3 + 512 + j];
                float xn = (float)XG[row * H3 + 1024 + j];
                float rg = sigm(xr + aR0[r] + aR1[r] + bhr);
                float zg = sigm(xz + aZ0[r] + aZ1[r] + bhz);
                float ng = tanhf(xn + rg * (aN0[r] + aN1[r] + bhn));
                int off = ((((m >> 4) * 16 + kk2i) * 32 + (m & 15) + lp) << 4) + e;
                float hp = (float)cur[off];
                float hnew = (1.f - zg) * ng + zg * hp;
                nxt[off] = (half_t)hnew;
                Hseq[row * HDIM + j] = (half_t)hnew;
            }
        }
        __syncthreads();
    }
}

// ---------------------------------------------------------------------------
// AttFlat softmax + weighted sum. One block (128 threads) per sequence.
// ---------------------------------------------------------------------------
__global__ void attflat_softmax_kernel(const float* __restrict__ a_raw,
                                       const half_t* __restrict__ Hseq,
                                       const int* __restrict__ que_ix,
                                       const int* __restrict__ opt_ix,
                                       const int* __restrict__ cp_ix,
                                       half_t* __restrict__ flat_sum_h) {
    int sq = blockIdx.x;           // 0..3711
    int T, rowBase;
    const int* toks;
    if (sq < 64)       { T = 64; rowBase = sq * 64;                toks = que_ix + sq * 64; }
    else if (sq < 512) { T = 16; rowBase = ROWO + (sq - 64) * 16;  toks = opt_ix + (sq - 64) * 16; }
    else               { T = 32; rowBase = ROWC + (sq - 512) * 32; toks = cp_ix + (sq - 512) * 32; }

    __shared__ float ps[64];
    __shared__ float wgt[64];
    __shared__ float st2[2];
    int tid = threadIdx.x;
    if (tid < T) ps[tid] = (toks[tid] == 0) ? NEGV : a_raw[rowBase + tid];
    __syncthreads();
    if (tid == 0) {
        float mx = -3.4e38f;
        for (int t = 0; t < T; ++t) mx = fmaxf(mx, ps[t]);
        float s = 0.f;
        for (int t = 0; t < T; ++t) s += expf(ps[t] - mx);
        st2[0] = mx; st2[1] = s;
    }
    __syncthreads();
    if (tid < T) wgt[tid] = expf(ps[tid] - st2[0]) / st2[1];
    __syncthreads();
    for (int h = tid; h < HDIM; h += blockDim.x) {
        float acc = 0.f;
        for (int t = 0; t < T; ++t)
            acc += wgt[t] * (float)Hseq[(size_t)(rowBase + t) * HDIM + h];
        flat_sum_h[(size_t)sq * HDIM + h] = (half_t)acc;
    }
}

__global__ void que_ent_kernel(const float* __restrict__ que_score,
                               const float* __restrict__ score_b2,
                               float* __restrict__ que_ent) {
    int b = threadIdx.x;
    if (b < BN) que_ent[b] = entf(sigm(que_score[b] + score_b2[0]));
}

// span_p [B,150,512] and qe [B,150,512] computed directly from cp_feat/que_vec
__global__ void span_pool_kernel(const float* __restrict__ feats_f,
                                 float* __restrict__ span_p,
                                 half_t* __restrict__ span_p_h,
                                 half_t* __restrict__ qe_h) {
    size_t id = (size_t)blockIdx.x * blockDim.x + threadIdx.x;
    if (id >= (size_t)BSN * HDIM) return;
    int p = (int)(id & 511);
    size_t t2 = id >> 9;
    int i = (int)(t2 % SPN);
    int b = (int)(t2 / SPN);
    int stc = i / WN, wi = i % WN;
    const float* cp_feat = feats_f + (size_t)512 * HDIM;  // cp rows start at seq 512
    const float* que_vec = feats_f;                        // que rows 0..63
    auto spanc = [&](int c) -> float {
        int off = c >> 9, h = c & 511;
        if (off <= wi && (stc + off) < SN)
            return cp_feat[(size_t)(b * SN + stc + off) * HDIM + h];
        return 0.f;
    };
    float sp = (spanc(3 * p) + spanc(3 * p + 1) + spanc(3 * p + 2)) * (1.f / 3.f);
    span_p[id] = sp;
    span_p_h[id] = (half_t)sp;
    float qv = 0.f;
    for (int q = 0; q < 4; ++q) {
        int c = 4 * p + q;
        qv += (c < HDIM) ? que_vec[(size_t)b * HDIM + c] : spanc(c - HDIM);
    }
    qe_h[id] = (half_t)(qv * 0.25f);
}

__global__ void empties_kernel(const int* __restrict__ opt_ix,
                               const int* __restrict__ cp_ix,
                               int* __restrict__ opt_empty,
                               int* __restrict__ sent_empty) {
    int id = blockIdx.x * blockDim.x + threadIdx.x;
    if (id < BN * AN) {
        int all0 = 1;
        for (int l = 0; l < LO; ++l) all0 &= (opt_ix[id * LO + l] == 0);
        opt_empty[id] = all0;
    } else if (id < BN * AN + BN * SN) {
        int sid = id - BN * AN;
        int all0 = 1;
        for (int l = 0; l < LS; ++l) all0 &= (cp_ix[sid * LS + l] == 0);
        sent_empty[sid] = all0;
    }
}

__global__ void evi_select_kernel(const float* __restrict__ sp_score,
                                  const float* __restrict__ qe_score,
                                  const float* __restrict__ que_ent,
                                  const float* __restrict__ score_b2,
                                  const int* __restrict__ sent_empty,
                                  int* __restrict__ evi_ix) {
    int b = threadIdx.x;
    if (b >= BN) return;
    float qent = que_ent[b];
    float b2 = score_b2[0];
    float best = -3.4e38f;
    int bix = 0;
    for (int i = 0; i < SPN; ++i) {
        float ig;
        if (sent_empty[b * SN + i / WN]) ig = NEGV;
        else {
            float el = sigm(sp_score[b * SPN + i] + b2);
            ig = qent - el * entf(sigm(qe_score[b * SPN + i] + b2));
        }
        if (ig > best) { best = ig; bix = i; }
    }
    evi_ix[b] = bix;
}

__global__ void evi_gather_kernel(const float* __restrict__ span_p,
                                  const int* __restrict__ evi_ix,
                                  float* __restrict__ evi_f,
                                  half_t* __restrict__ evi_h) {
    int id = blockIdx.x * blockDim.x + threadIdx.x;
    if (id >= BN * HDIM) return;
    int h = id & 511, b = id >> 9;
    float v = span_p[(size_t)(b * SPN + evi_ix[b]) * HDIM + h];
    evi_f[id] = v;
    evi_h[id] = (half_t)v;
}

__global__ void wmul_kernel(const float* __restrict__ v_f,
                            const float* __restrict__ ba_hmat,
                            half_t* __restrict__ w_h) {
    int id = blockIdx.x * blockDim.x + threadIdx.x;
    if (id >= BN * 2 * HDIM) return;
    int k = id & 511, g = (id >> 9) & 1, b = id >> 10;
    w_h[id] = (half_t)(v_f[b * HDIM + k] * ba_hmat[g * HDIM + k]);
}

__global__ void logits_kernel(const half_t* __restrict__ w_h,
                              const half_t* __restrict__ q_h,
                              const float* __restrict__ ba_hbias,
                              float* __restrict__ logits) {
    int id = blockIdx.x * blockDim.x + threadIdx.x;
    if (id >= BN * 2 * LQ) return;
    int qt = id & 63, g = (id >> 6) & 1, b = id >> 7;
    float acc = 0.f;
    for (int k = 0; k < HDIM; ++k)
        acc += (float)w_h[(size_t)(b * 2 + g) * HDIM + k] *
               (float)q_h[(size_t)(b * LQ + qt) * HDIM + k];
    logits[id] = acc + ba_hbias[g];
}

__global__ void attsm_kernel(const float* __restrict__ logits, float* __restrict__ att) {
    int id = threadIdx.x;   // b*2+g, 128 threads
    if (id >= BN * 2) return;
    const float* L = logits + id * LQ;
    float mx = -3.4e38f;
    for (int q = 0; q < LQ; ++q) mx = fmaxf(mx, L[q]);
    float s = 0.f;
    for (int q = 0; q < LQ; ++q) s += expf(L[q] - mx);
    for (int q = 0; q < LQ; ++q) att[id * LQ + q] = expf(L[q] - mx) / s;
}

__global__ void qinit_kernel(const half_t* __restrict__ Hseq,
                             float* __restrict__ qcur_f, half_t* __restrict__ qcur_h) {
    size_t id = (size_t)blockIdx.x * blockDim.x + threadIdx.x;
    if (id >= (size_t)BN * LQ * HDIM) return;
    half_t v = Hseq[id];   // que rows are the first 4096 rows
    qcur_f[id] = (float)v;
    qcur_h[id] = v;
}

__global__ void tk_kernel(const float* __restrict__ att,
                          const float* __restrict__ qg_f,
                          const float* __restrict__ vg_f,
                          half_t* __restrict__ bi_h, int g) {
    int id = blockIdx.x * blockDim.x + threadIdx.x;
    if (id >= BN * HDIM) return;
    int k = id & 511, b = id >> 9;
    float tacc = 0.f;
    for (int q = 0; q < LQ; ++q)
        tacc += att[(b * 2 + g) * LQ + q] * qg_f[(size_t)(b * LQ + q) * HDIM + k];
    bi_h[id] = (half_t)(vg_f[id] * tacc);
}

__global__ void qupd_kernel(const float* __restrict__ delta,
                            float* __restrict__ qcur_f, half_t* __restrict__ qcur_h) {
    size_t id = (size_t)blockIdx.x * blockDim.x + threadIdx.x;
    if (id >= (size_t)BN * LQ * HDIM) return;
    int k = (int)(id & 511);
    int b = (int)(id >> 9) >> 6;
    float v = qcur_f[id] + delta[b * HDIM + k];
    qcur_f[id] = v;
    qcur_h[id] = (half_t)v;
}

__global__ void fq_kernel(const float* __restrict__ qcur_f, float* __restrict__ fq) {
    int id = blockIdx.x * blockDim.x + threadIdx.x;
    if (id >= BN * HDIM) return;
    int k = id & 511, b = id >> 9;
    float acc = 0.f;
    for (int t = 0; t < LQ; ++t) acc += qcur_f[(size_t)(b * LQ + t) * HDIM + k];
    fq[id] = acc * (1.f / LQ);
}

__global__ void fuse_kernel(const float* __restrict__ feats_f,
                            const float* __restrict__ evi_f,
                            const float* __restrict__ fq,
                            half_t* __restrict__ fuse_h) {
    size_t id = (size_t)blockIdx.x * blockDim.x + threadIdx.x;
    if (id >= (size_t)BN * AN * 4096) return;
    int c = (int)(id & 4095);
    size_t row = id >> 12;               // b*7+a
    int b = (int)(row / AN);
    int sec = c >> 9, h = c & 511;
    float qv = feats_f[(size_t)b * HDIM + h];            // que_vec
    float ov = feats_f[(size_t)(64 + row) * HDIM + h];   // opt_feat
    float ev = evi_f[(size_t)b * HDIM + h];
    float fv = fq[(size_t)b * HDIM + h];
    float val = 0.f;
    switch (sec) {
        case 0: val = qv; break;
        case 1: val = ov; break;
        case 2: val = ev; break;
        case 3: val = qv * ev; break;
        case 4: val = ev * ov; break;
        case 5: val = qv * ov; break;
        case 6: val = fv; break;
        case 7: val = qv * ov * fv; break;
    }
    fuse_h[id] = (half_t)val;
}

__global__ void final_kernel(const float* __restrict__ proj_raw,
                             const float* __restrict__ cls_b2,
                             const int* __restrict__ opt_empty,
                             float* __restrict__ out) {
    int b = threadIdx.x;
    if (b >= BN) return;
    int cnt = 0;
    float b2 = cls_b2[0];
    for (int a = 0; a < AN; ++a) {
        int e = opt_empty[b * AN + a];
        out[b * AN + a] = e ? NEGV : (proj_raw[b * AN + a] + b2);
        cnt += !e;
    }
    out[BN * AN + b] = (float)cnt;
}

// ---------------------------------------------------------------------------
// Host orchestration
// ---------------------------------------------------------------------------
static inline unsigned g1(size_t n) { return (unsigned)((n + 255) / 256); }
static inline unsigned gemm_blocks(int M, int N) {
    long long mtg = ((M >> 4) + 7) / 8;
    return (unsigned)(mtg * (N >> 4));
}

extern "C" void kernel_launch(void* const* d_in, const int* in_sizes, int n_in,
                              void* d_out, int out_size, void* d_ws, size_t ws_size,
                              hipStream_t stream) {
    (void)in_sizes; (void)n_in; (void)out_size; (void)ws_size;
    const int*   que_ix   = (const int*)d_in[0];
    const int*   opt_ix   = (const int*)d_in[1];
    const int*   cp_ix    = (const int*)d_in[2];
    const float* emb      = (const float*)d_in[3];
    const float* gru_wi   = (const float*)d_in[4];
    const float* gru_wh   = (const float*)d_in[5];
    const float* gru_bi   = (const float*)d_in[6];
    const float* gru_bh   = (const float*)d_in[7];
    const float* flat_w1  = (const float*)d_in[8];
    const float* flat_b1  = (const float*)d_in[9];
    const float* flat_w2  = (const float*)d_in[10];
    const float* flat_mw  = (const float*)d_in[12];
    const float* flat_mb  = (const float*)d_in[13];
    const float* score_w1 = (const float*)d_in[14];
    const float* score_b1 = (const float*)d_in[15];
    const float* score_w2 = (const float*)d_in[16];
    const float* score_b2 = (const float*)d_in[17];
    const float* ba_vw    = (const float*)d_in[18];
    const float* ba_vb    = (const float*)d_in[19];
    const float* ba_qw    = (const float*)d_in[20];
    const float* ba_qb    = (const float*)d_in[21];
    const float* ba_hmat  = (const float*)d_in[22];
    const float* ba_hbias = (const float*)d_in[23];
    const float* bc_vw    = (const float*)d_in[24];
    const float* bc_vb    = (const float*)d_in[25];
    const float* bc_qw    = (const float*)d_in[26];
    const float* bc_qb    = (const float*)d_in[27];
    const float* prj_w    = (const float*)d_in[28];
    const float* prj_b    = (const float*)d_in[29];
    const float* cls_w1   = (const float*)d_in[30];
    const float* cls_b1   = (const float*)d_in[31];
    const float* cls_w2   = (const float*)d_in[32];
    const float* cls_b2   = (const float*)d_in[33];
    float* out = (float*)d_out;

    char* ws = (char*)d_ws;
    size_t off = 0;
    auto alloc = [&](size_t bytes) -> char* {
        off = (off + 255) & ~(size_t)255;
        char* p = ws + off;
        off += bytes;
        return p;
    };

    const size_t PACK512 = (size_t)32 * 16 * 512 * 2;   // K=512,N=512 pack
    half_t* Xh        = (half_t*)alloc((size_t)NTOK * EPAD * 2);
    half_t* WiF       = (half_t*)alloc((size_t)96 * 10 * 512 * 2);
    half_t* WhF       = (half_t*)alloc((size_t)96 * 16 * 512 * 2);
    half_t* w1F       = (half_t*)alloc(PACK512);
    half_t* mwF       = (half_t*)alloc(PACK512);
    half_t* sw1F      = (half_t*)alloc(PACK512);
    half_t* bavF      = (half_t*)alloc(PACK512);
    half_t* baqF      = (half_t*)alloc(PACK512);
    half_t* bcvF[2]   = { (half_t*)alloc(PACK512), (half_t*)alloc(PACK512) };
    half_t* bcqF[2]   = { (half_t*)alloc(PACK512), (half_t*)alloc(PACK512) };
    half_t* prjF[2]   = { (half_t*)alloc(PACK512), (half_t*)alloc(PACK512) };
    half_t* clsF      = (half_t*)alloc((size_t)32 * 128 * 512 * 2);
    half_t* XGh       = (half_t*)alloc((size_t)NTOK * H3 * 2);
    half_t* Hseq      = (half_t*)alloc((size_t)NTOK * HDIM * 2);
    float*  a_raw     = (float*)alloc((size_t)NTOK * 4);
    half_t* flat_sum  = (half_t*)alloc((size_t)NSEQ * HDIM * 2);
    float*  feats_f   = (float*)alloc((size_t)NSEQ * HDIM * 4);
    half_t* feats_h   = (half_t*)alloc((size_t)NSEQ * HDIM * 2);
    float*  que_score = (float*)alloc(BN * 4);
    float*  que_ent   = (float*)alloc(BN * 4);
    float*  span_p    = (float*)alloc((size_t)BSN * HDIM * 4);
    half_t* span_p_h  = (half_t*)alloc((size_t)BSN * HDIM * 2);
    half_t* qe_h      = (half_t*)alloc((size_t)BSN * HDIM * 2);
    float*  sp_score  = (float*)alloc(BSN * 4);
    float*  qe_score  = (float*)alloc(BSN * 4);
    int*    opt_empty = (int*)alloc(BN * AN * 4);
    int*    sent_empty= (int*)alloc(BN * SN * 4);
    int*    evi_ix    = (int*)alloc(BN * 4);
    float*  evi_f     = (float*)alloc(BN * HDIM * 4);
    half_t* evi_h     = (half_t*)alloc(BN * HDIM * 2);
    float*  v_f       = (float*)alloc(BN * HDIM * 4);
    half_t* v_h       = (half_t*)alloc(BN * HDIM * 2);
    float*  q__f      = (float*)alloc((size_t)BN * LQ * HDIM * 4);
    half_t* q__h      = (half_t*)alloc((size_t)BN * LQ * HDIM * 2);
    half_t* w_h       = (half_t*)alloc(BN * 2 * HDIM * 2);
    float*  logits    = (float*)alloc(BN * 2 * LQ * 4);
    float*  att       = (float*)alloc(BN * 2 * LQ * 4);
    float*  qcur_f    = (float*)alloc((size_t)BN * LQ * HDIM * 4);
    half_t* qcur_h    = (half_t*)alloc((size_t)BN * LQ * HDIM * 2);
    float*  vg_f      = (float*)alloc(BN * HDIM * 4);
    half_t* vg_h      = (half_t*)alloc(BN * HDIM * 2);
    float*  qg_f      = (float*)alloc((size_t)BN * LQ * HDIM * 4);
    half_t* qg_h      = (half_t*)alloc((size_t)BN * LQ * HDIM * 2);
    half_t* bi_h      = (half_t*)alloc(BN * HDIM * 2);
    float*  delta     = (float*)alloc(BN * HDIM * 4);
    float*  fq        = (float*)alloc(BN * HDIM * 4);
    half_t* fuse_h    = (half_t*)alloc((size_t)BN * AN * 4096 * 2);
    float*  proj_raw  = (float*)alloc(BN * AN * 4);

    // ---- pack weights to f16 fragment layout ----
    pack_b_kernel<<<g1((size_t)96 * 10 * 512), 256, 0, stream>>>(gru_wi, WiF, 300, EPAD, H3, 1);
    pack_b_kernel<<<g1((size_t)96 * 16 * 512), 256, 0, stream>>>(gru_wh, WhF, 512, 512, H3, 1);
    pack_b_kernel<<<g1((size_t)32 * 16 * 512), 256, 0, stream>>>(flat_w1, w1F, 512, 512, 512, 0);
    pack_b_kernel<<<g1((size_t)32 * 16 * 512), 256, 0, stream>>>(flat_mw, mwF, 512, 512, 512, 0);
    pack_b_kernel<<<g1((size_t)32 * 16 * 512), 256, 0, stream>>>(score_w1, sw1F, 512, 512, 512, 0);
    pack_b_kernel<<<g1((size_t)32 * 16 * 512), 256, 0, stream>>>(ba_vw, bavF, 512, 512, 512, 0);
    pack_b_kernel<<<g1((size_t)32 * 16 * 512), 256, 0, stream>>>(ba_qw, baqF, 512, 512, 512, 0);
    for (int g = 0; g < 2; ++g) {
        pack_b_kernel<<<g1((size_t)32 * 16 * 512), 256, 0, stream>>>(bc_vw + (size_t)g * 512 * 512, bcvF[g], 512, 512, 512, 0);
        pack_b_kernel<<<g1((size_t)32 * 16 * 512), 256, 0, stream>>>(bc_qw + (size_t)g * 512 * 512, bcqF[g], 512, 512, 512, 0);
        pack_b_kernel<<<g1((size_t)32 * 16 * 512), 256, 0, stream>>>(prj_w + (size_t)g * 512 * 512, prjF[g], 512, 512, 512, 0);
    }
    pack_b_kernel<<<g1((size_t)32 * 128 * 512), 256, 0, stream>>>(cls_w1, clsF, 4096, 4096, 512, 0);

    // ---- embeddings ----
    gather_embed_kernel<<<g1((size_t)NTOK * EPAD), 256, 0, stream>>>(que_ix, opt_ix, cp_ix, emb, Xh);

    // ---- GRU input projection: XG = X @ Wi^T + bi  (f16 out) ----
    gemm_wmma<0, false><<<gemm_blocks(NTOK, H3), 256, 0, stream>>>(
        Xh, EPAD, WiF, gru_bi, NTOK, H3, EPAD, nullptr, XGh, nullptr, nullptr);

    // ---- GRU recurrent scan (116 WGs, 64KB dynamic LDS) ----
    gru_scan_kernel<<<116, 256, 65536, stream>>>(XGh, WhF, gru_bh, Hseq);

    // ---- AttFlat token scores: a = relu(h@w1+b1)@w2 (fused, atomic) ----
    (void)hipMemsetAsync(a_raw, 0, (size_t)NTOK * 4, stream);
    gemm_wmma<3, true><<<gemm_blocks(NTOK, HDIM), 256, 0, stream>>>(
        Hseq, HDIM, w1F, flat_b1, NTOK, HDIM, HDIM, nullptr, nullptr, flat_w2, a_raw);
    attflat_softmax_kernel<<<NSEQ, 128, 0, stream>>>(a_raw, Hseq, que_ix, opt_ix, cp_ix, flat_sum);
    gemm_wmma<2, false><<<gemm_blocks(NSEQ, HDIM), 256, 0, stream>>>(
        flat_sum, HDIM, mwF, flat_mb, NSEQ, HDIM, HDIM, feats_f, feats_h, nullptr, nullptr);

    // ---- que entropy ----
    (void)hipMemsetAsync(que_score, 0, BN * 4, stream);
    gemm_wmma<3, false><<<gemm_blocks(BN, HDIM), 256, 0, stream>>>(
        feats_h, HDIM, sw1F, score_b1, BN, HDIM, HDIM, nullptr, nullptr, score_w2, que_score);
    que_ent_kernel<<<1, 64, 0, stream>>>(que_score, score_b2, que_ent);

    // ---- span features + pooled scores ----
    span_pool_kernel<<<g1((size_t)BSN * HDIM), 256, 0, stream>>>(feats_f, span_p, span_p_h, qe_h);
    (void)hipMemsetAsync(sp_score, 0, BSN * 4, stream);
    (void)hipMemsetAsync(qe_score, 0, BSN * 4, stream);
    gemm_wmma<3, false><<<gemm_blocks(BSN, HDIM), 256, 0, stream>>>(
        span_p_h, HDIM, sw1F, score_b1, BSN, HDIM, HDIM, nullptr, nullptr, score_w2, sp_score);
    gemm_wmma<3, false><<<gemm_blocks(BSN, HDIM), 256, 0, stream>>>(
        qe_h, HDIM, sw1F, score_b1, BSN, HDIM, HDIM, nullptr, nullptr, score_w2, qe_score);

    empties_kernel<<<g1(BN * AN + BN * SN), 256, 0, stream>>>(opt_ix, cp_ix, opt_empty, sent_empty);
    evi_select_kernel<<<1, 64, 0, stream>>>(sp_score, qe_score, que_ent, score_b2, sent_empty, evi_ix);
    evi_gather_kernel<<<g1(BN * HDIM), 256, 0, stream>>>(span_p, evi_ix, evi_f, evi_h);

    // ---- bilinear attention ----
    gemm_wmma<2, true><<<gemm_blocks(BN, HDIM), 256, 0, stream>>>(
        evi_h, HDIM, bavF, ba_vb, BN, HDIM, HDIM, v_f, v_h, nullptr, nullptr);
    gemm_wmma<2, true><<<gemm_blocks(BN * LQ, HDIM), 256, 0, stream>>>(
        Hseq, HDIM, baqF, ba_qb, BN * LQ, HDIM, HDIM, q__f, q__h, nullptr, nullptr);
    wmul_kernel<<<g1(BN * 2 * HDIM), 256, 0, stream>>>(v_f, ba_hmat, w_h);
    logits_kernel<<<g1(BN * 2 * LQ), 256, 0, stream>>>(w_h, q__h, ba_hbias, logits);
    attsm_kernel<<<1, 128, 0, stream>>>(logits, att);
    qinit_kernel<<<g1((size_t)BN * LQ * HDIM), 256, 0, stream>>>(Hseq, qcur_f, qcur_h);

    for (int g = 0; g < 2; ++g) {
        gemm_wmma<2, true><<<gemm_blocks(BN, HDIM), 256, 0, stream>>>(
            evi_h, HDIM, bcvF[g], bc_vb + g * HDIM, BN, HDIM, HDIM, vg_f, vg_h, nullptr, nullptr);
        gemm_wmma<2, true><<<gemm_blocks(BN * LQ, HDIM), 256, 0, stream>>>(
            qcur_h, HDIM, bcqF[g], bc_qb + g * HDIM, BN * LQ, HDIM, HDIM, qg_f, qg_h, nullptr, nullptr);
        tk_kernel<<<g1(BN * HDIM), 256, 0, stream>>>(att, qg_f, vg_f, bi_h, g);
        gemm_wmma<1, false><<<gemm_blocks(BN, HDIM), 256, 0, stream>>>(
            bi_h, HDIM, prjF[g], prj_b + g * HDIM, BN, HDIM, HDIM, delta, nullptr, nullptr, nullptr);
        qupd_kernel<<<g1((size_t)BN * LQ * HDIM), 256, 0, stream>>>(delta, qcur_f, qcur_h);
    }
    fq_kernel<<<g1(BN * HDIM), 256, 0, stream>>>(qcur_f, fq);

    // ---- classifier ----
    fuse_kernel<<<g1((size_t)BN * AN * 4096), 256, 0, stream>>>(feats_f, evi_f, fq, fuse_h);
    (void)hipMemsetAsync(proj_raw, 0, BN * AN * 4, stream);
    gemm_wmma<3, true><<<gemm_blocks(BN * AN, HDIM), 256, 0, stream>>>(
        fuse_h, 4096, clsF, cls_b1, BN * AN, HDIM, 4096, nullptr, nullptr, cls_w2, proj_raw);
    final_kernel<<<1, 64, 0, stream>>>(proj_raw, cls_b2, opt_empty, out);
}
